// LSTM_58042188038726
// MI455X (gfx1250) — matrix operand order repaired and checked by reading, not compiled, not run on hardware
//
#include <hip/hip_runtime.h>

typedef __attribute__((ext_vector_type(16))) _Float16 v16h;
typedef __attribute__((ext_vector_type(8)))  _Float16 v8h;
typedef __attribute__((ext_vector_type(8)))  float    v8f;

// Global (device memory) address-space views — force global_load_b128
// (LOADcnt only) instead of flat_load_b128 (LOADcnt+DScnt) for pointers
// whose provenance we deliberately hide from LICM.
typedef __attribute__((address_space(1))) const _Float16 gh_t;
typedef __attribute__((address_space(1))) const v8h     gv8h_t;

#define INPUT_SIZE 64
#define HIDDEN     256
#define BATCH      256
#define SEQ        512
#define KDIM       (INPUT_SIZE + HIDDEN)   // 320 = fused [x | h] K dimension
#define NKT        (KDIM / 32)             // 10 K-tiles of 32
#define NKT_REG    4                       // K-tiles whose B frags live in VGPRs
#define NNT        ((4 * HIDDEN) / 16)     // 64 N-tiles of 16
#define BLOB_HALFS 512                     // one WMMA B fragment: 32 lanes x 16 halfs

// ---------------------------------------------------------------------------
// Pack [W_ih ; W_hh] (K-stacked, f32) into f16 WMMA-B fragment layout.
// B is 32x16 (KxN) per fragment; lane L (0-15) holds column N=L, K=0..15
// (2 per VGPR); lane L (16-31) holds column N=L-16, K=16..31.
// Blob (nt, kt) lives at Wc[(nt*NKT + kt)*512 .. +512).
// ---------------------------------------------------------------------------
__global__ __launch_bounds__(512) void lstm_pack_w(const float* __restrict__ W_ih,
                                                   const float* __restrict__ W_hh,
                                                   _Float16* __restrict__ Wc) {
  int idx    = blockIdx.x * 512 + threadIdx.x;   // 0 .. 640*512-1
  int blob   = idx >> 9;
  int within = idx & 511;
  int lane   = within >> 4;
  int j      = within & 15;
  int nt = blob / NKT;
  int kt = blob % NKT;
  int n = nt * 16 + (lane & 15);                 // gate/output column (0..1023)
  int k = kt * 32 + ((lane < 16) ? j : 16 + j);  // fused K index (0..319)
  float v = (k < INPUT_SIZE) ? W_ih[n * INPUT_SIZE + k]
                             : W_hh[n * HIDDEN + (k - INPUT_SIZE)];
  Wc[idx] = (_Float16)v;
}

__device__ __forceinline__ float sigm(float x) { return 1.0f / (1.0f + __expf(-x)); }

__device__ __forceinline__ v16h load_frag16(const _Float16* p) {
  v8h q0 = *(const v8h*)p;
  v8h q1 = *(const v8h*)(p + 8);
  v16h r;
  #pragma unroll
  for (int q = 0; q < 8; ++q) { r[q] = q0[q]; r[8 + q] = q1[q]; }
  return r;
}

__device__ __forceinline__ v16h load_frag16_g(gh_t* p) {
  v8h q0 = *(gv8h_t*)p;
  v8h q1 = *(gv8h_t*)(p + 8);
  v16h r;
  #pragma unroll
  for (int q = 0; q < 8; ++q) { r[q] = q0[q]; r[8 + q] = q1[q]; }
  return r;
}

__device__ __forceinline__ v8f wmma16(v16h a, v16h b, v8f c) {
  return __builtin_amdgcn_wmma_f32_16x16x32_f16(false, a, false, b,
                                                (short)0, c, false, false);
}

// ---------------------------------------------------------------------------
// One block = one 16-row batch tile for the whole sequence.
// 16 waves; wave w computes N-tiles {w, w+16, w+32, w+48} = i/f/g/o gates for
// the SAME 16 hidden columns -> pointwise LSTM update stays in registers.
// c state lives in VGPRs for all 512 steps; only h goes to LDS (next A).
// B fragments: K-tiles 0..3 pinned in VGPRs, K-tiles 4..9 streamed from L2.
// ---------------------------------------------------------------------------
__global__ __launch_bounds__(512, 1) void lstm_seq(
    const float*    __restrict__ x,      // [B, T, I] f32
    const _Float16* __restrict__ Wc,     // packed B fragments
    const float*    __restrict__ b_ih,   // [4H]
    const float*    __restrict__ b_hh,   // [4H]
    const float*    __restrict__ W_fc,   // [1, H]
    const float*    __restrict__ b_fc,   // [1]
    float*          __restrict__ out)    // [B, 1]
{
  __shared__ __align__(16) _Float16 a_sh[16][KDIM];  // A-stage: [x_t | h], 10 KB

  const int tid  = threadIdx.x;
  const int lane = tid & 31;
  const int wv   = tid >> 5;        // wave id 0..15
  const int hi   = lane >> 4;       // half-wave: 0 or 1
  const int rowm = lane & 15;       // A-matrix row M this lane reads
  const int hcol = wv * 16 + rowm;  // hidden column this lane owns in C/D
  const int b0   = blockIdx.x * 16; // batch-tile base row
  const int c0   = hi * 16;         // A K-chunk offset for this half-wave

  // Per-lane gate biases (fixed column per lane for the whole sequence).
  const float bi = b_ih[hcol]       + b_hh[hcol];
  const float bf = b_ih[256 + hcol] + b_hh[256 + hcol];
  const float bg = b_ih[512 + hcol] + b_hh[512 + hcol];
  const float bo = b_ih[768 + hcol] + b_hh[768 + hcol];

  // Pin B fragments for K-tiles 0..NKT_REG-1 (all 4 gates) in registers.
  v16h wB[4 * NKT_REG];
  {
    const _Float16* bbase = Wc + (size_t)lane * 16;
    #pragma unroll
    for (int g = 0; g < 4; ++g)
      #pragma unroll
      for (int kt = 0; kt < NKT_REG; ++kt)
        wB[g * NKT_REG + kt] =
            load_frag16(bbase + ((size_t)(wv + 16 * g) * NKT + kt) * BLOB_HALFS);
  }

  // Init: h0 = 0 in LDS, stage x_0.
  for (int e = tid; e < 16 * HIDDEN; e += 512)
    a_sh[e >> 8][INPUT_SIZE + (e & 255)] = (_Float16)0.0f;
  for (int e = tid; e < 16 * INPUT_SIZE; e += 512) {
    int b = e >> 6, i = e & 63;
    a_sh[b][i] = (_Float16)x[((size_t)(b0 + b) * SEQ) * INPUT_SIZE + i];
  }

  float c_reg[8];
  #pragma unroll
  for (int r = 0; r < 8; ++r) c_reg[r] = 0.0f;

  // LICM-opaque view of the streamed weight pointer (prevents the compiler
  // from hoisting all streamed B loads out of the t-loop and spilling them).
  uintptr_t wptr = (uintptr_t)Wc;

  __syncthreads();

  #pragma unroll 1
  for (int t = 0; t < SEQ; ++t) {
    asm volatile("" : "+s"(wptr));   // re-materialize each step: block LICM
    // Rebuild as an explicit global (AS1) pointer so streamed weight loads
    // lower to global_load_b128 (LOADcnt only), not flat_load (LOADcnt+DScnt).
    gh_t* Wc_dyn = (gh_t*)wptr;

    // Start x_{t+1} global loads early; convert/store after mid-barrier.
    float xr0 = 0.0f, xr1 = 0.0f;
    if (t + 1 < SEQ) {
      int e0 = tid, e1 = tid + 512;
      xr0 = x[((size_t)(b0 + (e0 >> 6)) * SEQ + (t + 1)) * INPUT_SIZE + (e0 & 63)];
      xr1 = x[((size_t)(b0 + (e1 >> 6)) * SEQ + (t + 1)) * INPUT_SIZE + (e1 & 63)];
    }

    v8f acc_i = {}, acc_f = {}, acc_g = {}, acc_o = {};

    // ---- K-tiles with register-resident B fragments.
    #pragma unroll
    for (int kt = 0; kt < NKT_REG; ++kt) {
      v16h afrag = load_frag16(&a_sh[rowm][kt * 32 + c0]);
      acc_i = wmma16(afrag, wB[0 * NKT_REG + kt], acc_i);
      acc_f = wmma16(afrag, wB[1 * NKT_REG + kt], acc_f);
      acc_g = wmma16(afrag, wB[2 * NKT_REG + kt], acc_g);
      acc_o = wmma16(afrag, wB[3 * NKT_REG + kt], acc_o);
    }
    // ---- K-tiles streamed from L2 (weights are permanently L2-resident).
    #pragma unroll
    for (int kt = NKT_REG; kt < NKT; ++kt) {
      v16h afrag = load_frag16(&a_sh[rowm][kt * 32 + c0]);
      gh_t* bb = Wc_dyn + (size_t)lane * 16;
      acc_i = wmma16(afrag, load_frag16_g(bb + ((size_t)(wv)      * NKT + kt) * BLOB_HALFS), acc_i);
      acc_f = wmma16(afrag, load_frag16_g(bb + ((size_t)(wv + 16) * NKT + kt) * BLOB_HALFS), acc_f);
      acc_g = wmma16(afrag, load_frag16_g(bb + ((size_t)(wv + 32) * NKT + kt) * BLOB_HALFS), acc_g);
      acc_o = wmma16(afrag, load_frag16_g(bb + ((size_t)(wv + 48) * NKT + kt) * BLOB_HALFS), acc_o);
    }

    __syncthreads();  // all waves done reading a_sh

    // Pointwise LSTM update, fully in registers.
    // C/D layout: VGPR r -> row (r + 8*hi), column hcol for this lane.
    #pragma unroll
    for (int r = 0; r < 8; ++r) {
      float gi = sigm(acc_i[r] + bi);
      float gf = sigm(acc_f[r] + bf);
      float gg = tanhf(acc_g[r] + bg);
      float go = sigm(acc_o[r] + bo);
      float cn = gf * c_reg[r] + gi * gg;
      c_reg[r] = cn;
      float hn = go * tanhf(cn);
      a_sh[r + 8 * hi][INPUT_SIZE + hcol] = (_Float16)hn;
    }

    // Stage x_{t+1} (loads already in flight since before the GEMM).
    if (t + 1 < SEQ) {
      a_sh[tid >> 6][tid & 63]         = (_Float16)xr0;
      a_sh[(tid + 512) >> 6][tid & 63] = (_Float16)xr1;
    }
    __syncthreads();
  }

  // Final FC: out[b] = h_final[b,:] . W_fc + b_fc  (PRED_OUT = 1)
  if (tid < 16) {
    float s = b_fc[0];
    for (int k = 0; k < HIDDEN; ++k)
      s += (float)a_sh[tid][INPUT_SIZE + k] * W_fc[k];
    out[b0 + tid] = s;
  }
}

extern "C" void kernel_launch(void* const* d_in, const int* in_sizes, int n_in,
                              void* d_out, int out_size, void* d_ws, size_t ws_size,
                              hipStream_t stream) {
  (void)in_sizes; (void)n_in; (void)out_size; (void)ws_size;
  const float* x    = (const float*)d_in[0];
  const float* W_ih = (const float*)d_in[1];
  const float* W_hh = (const float*)d_in[2];
  const float* b_ih = (const float*)d_in[3];
  const float* b_hh = (const float*)d_in[4];
  const float* W_fc = (const float*)d_in[5];
  const float* b_fc = (const float*)d_in[6];
  float* out = (float*)d_out;
  _Float16* Wc = (_Float16*)d_ws;  // 640 blobs * 512 halfs = 640 KB

  lstm_pack_w<<<NNT * NKT, 512, 0, stream>>>(W_ih, W_hh, Wc);
  lstm_seq<<<BATCH / 16, 512, 0, stream>>>(x, Wc, b_ih, b_hh, W_fc, b_fc, out);
}
